// custom_lstm_5420248728116
// MI455X (gfx1250) — matrix operand order repaired and checked by reading, not compiled, not run on hardware
//
#include <hip/hip_runtime.h>
#include <stdint.h>

// ---------------------------------------------------------------------------
// Bidirectional LSTM for MI455X (gfx1250, wave32, WMMA).
//   B=64, T=2048, D=256, H=256, gates G=4H=1024.
// Phase 0a: pack W^T, U^T to bf16 (B fragments become contiguous 32B rows).
// Phase 0b: convert x to bf16 ONCE (removes per-tile f32->bf16 VALU work).
// Phase 1 : xw = x @ W + bias via v_wmma_f32_16x16x32_bf16, stored in
//           fragment-major order so phase 2 reads it with one b128 per lane.
// Phase 2 : 2048 sequential steps, 16 WGs/direction, grid barrier per step.
// ---------------------------------------------------------------------------

#define B_   64
#define T_   2048
#define D_   256
#define H_   256
#define G_   1024   // 4H
#define NWG_ 16     // workgroups per direction in recurrence

typedef __attribute__((ext_vector_type(8)))  float    v8f;
typedef __attribute__((ext_vector_type(8)))  uint32_t v8u;
typedef __attribute__((ext_vector_type(16))) __bf16   v16bf;

__device__ __forceinline__ uint32_t f2bf_u(float f) {
    uint32_t u = __float_as_uint(f);
    u += 0x7FFFu + ((u >> 16) & 1u);     // round-to-nearest-even
    return u >> 16;
}
__device__ __forceinline__ uint16_t f2bf(float f) { return (uint16_t)f2bf_u(f); }
__device__ __forceinline__ uint32_t pack2(float lo, float hi) {
    return f2bf_u(lo) | (f2bf_u(hi) << 16);   // elem 2i in [15:0], 2i+1 in [31:16]
}
__device__ __forceinline__ float bf2f_lo(uint32_t w) {
    return __uint_as_float(w << 16);
}
__device__ __forceinline__ float bf2f_hi(uint32_t w) {
    return __uint_as_float(w & 0xFFFF0000u);
}

__device__ __forceinline__ v8f wmma_bf16(v8u a, v8u b, v8f c) {
    return __builtin_amdgcn_wmma_f32_16x16x32_bf16(
        false, __builtin_bit_cast(v16bf, a),
        false, __builtin_bit_cast(v16bf, b),
        (short)0, c, false, false);
}

__device__ __forceinline__ float sigm(float x) { return 1.0f / (1.0f + __expf(-x)); }

// ---------------------------------------------------------------------------
// Phase 0a: transpose+convert weights to bf16, zero h state.
// ---------------------------------------------------------------------------
__global__ __launch_bounds__(256) void init_pack(
    const float* __restrict__ Wf, const float* __restrict__ Uf,
    const float* __restrict__ Wb, const float* __restrict__ Ub,
    uint16_t* __restrict__ Wtf, uint16_t* __restrict__ Utf,
    uint16_t* __restrict__ Wtb, uint16_t* __restrict__ Utb,
    uint16_t* __restrict__ hF,  uint16_t* __restrict__ hB)
{
    int idx = blockIdx.x * 256 + threadIdx.x;    // [0, 262144)
    int d = idx & 255;
    int g = idx >> 8;                            // [0, 1024)
    Wtf[g * D_ + d] = f2bf(Wf[(size_t)d * G_ + g]);
    Utf[g * H_ + d] = f2bf(Uf[(size_t)d * G_ + g]);
    Wtb[g * D_ + d] = f2bf(Wb[(size_t)d * G_ + g]);
    Utb[g * H_ + d] = f2bf(Ub[(size_t)d * G_ + g]);
    if (idx < B_ * H_) { hF[idx] = 0; hB[idx] = 0; }   // h_0 = 0 (both dirs)
}

// ---------------------------------------------------------------------------
// Phase 0b: x (f32) -> xbf (bf16), one streaming pass, 8 elems/thread.
// ---------------------------------------------------------------------------
__global__ __launch_bounds__(256) void cvt_x(
    const float* __restrict__ x, uint16_t* __restrict__ xbf)
{
    size_t i = ((size_t)blockIdx.x * 256 + threadIdx.x) * 8;   // elem index
    const float4* p = reinterpret_cast<const float4*>(x + i);
    float4 a = p[0], b = p[1];
    uint4 o;
    o.x = pack2(a.x, a.y); o.y = pack2(a.z, a.w);
    o.z = pack2(b.x, b.y); o.w = pack2(b.z, b.w);
    *reinterpret_cast<uint4*>(xbf + i) = o;
}

// ---------------------------------------------------------------------------
// Phase 1: xw = x @ W + bias.  Block = 8 waves, one 16x16 tile per wave,
// K=256 (8 WMMA).  Waves in a block share A rows (L0/L1 reuse).
// Output layout (fragment-major): xw[t][tile][lane][r] bf16, where
// tile = gt*4 + mt, gt = gate-col tile [0,64), mt = batch-row tile [0,4).
// ---------------------------------------------------------------------------
__global__ __launch_bounds__(256) void xw_gemm(
    const uint16_t* __restrict__ xbf,
    const uint16_t* __restrict__ Wtf, const uint16_t* __restrict__ Wtb,
    const float* __restrict__ biasF,  const float* __restrict__ biasB,
    uint16_t* __restrict__ xwF, uint16_t* __restrict__ xwB)
{
    const int dir = blockIdx.y;
    const uint16_t* Wt  = dir ? Wtb : Wtf;
    const float*    bia = dir ? biasB : biasF;
    uint16_t*       xw  = dir ? xwB : xwF;

    const int bid  = blockIdx.x;           // [0, T*32)
    const int t    = bid >> 5;
    const int sub  = bid & 31;             // 32 blocks cover one t (256 tiles)
    const int wave = threadIdx.x >> 5;
    const int lane = threadIdx.x & 31;

    const int mt = sub & 3;                              // batch-row tile
    const int gt = (sub >> 2) * 8 + wave;                // gate-col tile [0,64)
    const int m0 = mt * 16;
    const int g0 = gt * 16;
    const int t_src = dir ? (T_ - 1 - t) : t;            // bwd dir scans reversed x

    // A fragment: row = m0 + lane%16 of xbf[:, t_src, :]
    const int arow = m0 + (lane & 15);
    const int ksel = (lane >> 4) * 16;                   // K-half select (elems)
    const uint16_t* xrow = xbf + ((size_t)arow * T_ + t_src) * D_;

    // B fragment: column g0 + lane%16 of W  (== row of Wt), 32B contiguous
    const int colB = g0 + (lane & 15);
    const int kb   = (lane >> 4) * 16;
    const uint16_t* brow = Wt + (size_t)colB * D_;

    v8f acc = {0.f, 0.f, 0.f, 0.f, 0.f, 0.f, 0.f, 0.f};

    #pragma unroll
    for (int k0 = 0; k0 < D_; k0 += 32) {
        uint4 a0 = *reinterpret_cast<const uint4*>(xrow + k0 + ksel);
        uint4 a1 = *reinterpret_cast<const uint4*>(xrow + k0 + 8 + ksel);
        v8u af;
        af[0] = a0.x; af[1] = a0.y; af[2] = a0.z; af[3] = a0.w;
        af[4] = a1.x; af[5] = a1.y; af[6] = a1.z; af[7] = a1.w;
        const uint4* bp = reinterpret_cast<const uint4*>(brow + k0 + kb);
        uint4 b0 = bp[0], b1 = bp[1];
        v8u bf;
        bf[0] = b0.x; bf[1] = b0.y; bf[2] = b0.z; bf[3] = b0.w;
        bf[4] = b1.x; bf[5] = b1.y; bf[6] = b1.z; bf[7] = b1.w;
        acc = wmma_bf16(af, bf, acc);
    }

    // bias is constant per lane (C layout: col = g0 + lane%16 for all r)
    const float bb = bia[g0 + (lane & 15)];
    const int tile = gt * 4 + mt;
    uint4 st;
    st.x = pack2(acc[0] + bb, acc[1] + bb);
    st.y = pack2(acc[2] + bb, acc[3] + bb);
    st.z = pack2(acc[4] + bb, acc[5] + bb);
    st.w = pack2(acc[6] + bb, acc[7] + bb);
    uint16_t* dst = xw + (((size_t)t * 256 + tile) * 256) + lane * 8;
    *reinterpret_cast<uint4*>(dst) = st;
}

// ---------------------------------------------------------------------------
// Phase 2: sequential recurrence. 32 blocks: dir = blk/16, wg = blk%16.
// Each WG owns h-slice j in [wg*16, wg*16+16): 16 waves = 4 batch-row tiles
// x 4 gates.  U fragments register-resident across the whole t-loop.
// ---------------------------------------------------------------------------
__global__ __launch_bounds__(512) void lstm_rec(
    const uint16_t* __restrict__ Utf, const uint16_t* __restrict__ Utb,
    const uint16_t* __restrict__ xwF, const uint16_t* __restrict__ xwB,
    uint16_t* __restrict__ hF, uint16_t* __restrict__ hB,
    uint32_t* ctr, float* __restrict__ out)
{
    const int dir = blockIdx.x >> 4;
    const int wg  = blockIdx.x & 15;
    const uint16_t* Ut  = dir ? Utb : Utf;
    const uint16_t* xw  = dir ? xwB : xwF;
    uint16_t*       hbf = dir ? hB  : hF;
    uint32_t*       cp  = ctr + dir;

    const int wave = threadIdx.x >> 5;
    const int lane = threadIdx.x & 31;
    const int m  = wave & 3;        // batch-row tile
    const int q  = wave >> 2;       // gate (i,f,g,o)
    const int j0 = wg * 16;
    const int m0 = m * 16;
    const int gcol0 = q * H_ + j0;

    __shared__ float zs[4][B_][16];   // gate pre-activations for this j-slice
    __shared__ float cs[B_][16];      // persistent cell state for this j-slice

    for (int i = threadIdx.x; i < B_ * 16; i += 512) ((float*)cs)[i] = 0.0f;
    __syncthreads();

    // Preload B fragments: U^T rows for our 16 gate columns, full K=256.
    const int colB = gcol0 + (lane & 15);
    const int kb   = (lane >> 4) * 16;
    v8u bfr[8];
    #pragma unroll
    for (int i = 0; i < 8; ++i) {
        const uint4* bp = reinterpret_cast<const uint4*>(
            Ut + (size_t)colB * H_ + i * 32 + kb);
        uint4 b0 = bp[0], b1 = bp[1];
        bfr[i][0] = b0.x; bfr[i][1] = b0.y; bfr[i][2] = b0.z; bfr[i][3] = b0.w;
        bfr[i][4] = b1.x; bfr[i][5] = b1.y; bfr[i][6] = b1.z; bfr[i][7] = b1.w;
    }

    const int arow = m0 + (lane & 15);
    const int ksel = (lane >> 4) * 16;
    const uint16_t* hrow = hbf + (size_t)arow * H_;
    // fragment-major xw tile for this wave: tile = (q*16 + wg)*4 + m
    const int tile = (q * 16 + wg) * 4 + m;
    const uint16_t* xwlane = xw + ((size_t)tile * 256) + lane * 8;

    for (int t = 0; t < T_; ++t) {
        // z_tile = h @ U   (K = 256 -> 8 WMMA)
        v8f acc = {0.f, 0.f, 0.f, 0.f, 0.f, 0.f, 0.f, 0.f};
        #pragma unroll
        for (int i = 0; i < 8; ++i) {
            const int k0 = i * 32;
            uint4 a0 = *reinterpret_cast<const uint4*>(hrow + k0 + ksel);
            uint4 a1 = *reinterpret_cast<const uint4*>(hrow + k0 + 8 + ksel);
            v8u af;
            af[0] = a0.x; af[1] = a0.y; af[2] = a0.z; af[3] = a0.w;
            af[4] = a1.x; af[5] = a1.y; af[6] = a1.z; af[7] = a1.w;
            acc = wmma_bf16(af, bfr[i], acc);
        }
        // z += xw[t] : one coalesced b128 per lane (fragment-major layout)
        uint4 xv = *reinterpret_cast<const uint4*>(xwlane + (size_t)t * (256 * 256));
        acc[0] += bf2f_lo(xv.x); acc[1] += bf2f_hi(xv.x);
        acc[2] += bf2f_lo(xv.y); acc[3] += bf2f_hi(xv.y);
        acc[4] += bf2f_lo(xv.z); acc[5] += bf2f_hi(xv.z);
        acc[6] += bf2f_lo(xv.w); acc[7] += bf2f_hi(xv.w);

        // scatter z to LDS (C layout: row = m0 + r + 8*(lane/16), col = lane%16)
        #pragma unroll
        for (int r = 0; r < 8; ++r) {
            int rr = m0 + r + 8 * (lane >> 4);
            zs[q][rr][lane & 15] = acc[r];
        }
        __syncthreads();

        // Elementwise gates: 64x16 elements, 512 threads x 2
        for (int e = threadIdx.x; e < B_ * 16; e += 512) {
            int b = e >> 4, j = e & 15;
            float ig = sigm(zs[0][b][j]);
            float fg = sigm(zs[1][b][j]);
            float gg = tanhf(zs[2][b][j]);
            float og = sigm(zs[3][b][j]);
            float cn = fg * cs[b][j] + ig * gg;
            cs[b][j] = cn;
            float hn = og * tanhf(cn);
            hbf[(size_t)b * H_ + j0 + j] = f2bf(hn);
            if (t == T_ - 1)
                out[(size_t)b * (2 * H_) + dir * H_ + j0 + j] = hn;
        }
        __threadfence();
        __syncthreads();

        // grid barrier across the 16 WGs of this direction (monotonic counter)
        if (threadIdx.x == 0) {
            __hip_atomic_fetch_add(cp, 1u, __ATOMIC_RELEASE, __HIP_MEMORY_SCOPE_AGENT);
            const uint32_t tgt = (uint32_t)NWG_ * (uint32_t)(t + 1);
            while (__hip_atomic_load(cp, __ATOMIC_ACQUIRE, __HIP_MEMORY_SCOPE_AGENT) < tgt)
                __builtin_amdgcn_s_sleep(2);
        }
        __syncthreads();
    }
}

// ---------------------------------------------------------------------------
// Workspace layout (bytes):
//   Utf 512K | Utb 512K | Wtf 512K | Wtb 512K | hF 32K | hB 32K | ctr 256
//   xbf 64M | xwF 256M | xwB 256M            total ~578 MB
// ---------------------------------------------------------------------------
extern "C" void kernel_launch(void* const* d_in, const int* in_sizes, int n_in,
                              void* d_out, int out_size, void* d_ws, size_t ws_size,
                              hipStream_t stream) {
    const float* x  = (const float*)d_in[0];
    const float* Wf = (const float*)d_in[1];
    const float* Uf = (const float*)d_in[2];
    const float* bf = (const float*)d_in[3];
    const float* Wb = (const float*)d_in[4];
    const float* Ub = (const float*)d_in[5];
    const float* bb = (const float*)d_in[6];
    float* out = (float*)d_out;

    char* ws = (char*)d_ws;
    const size_t SZ_WT = (size_t)G_ * D_ * sizeof(uint16_t);        // 512 KB
    const size_t SZ_H  = (size_t)B_ * H_ * sizeof(uint16_t);        // 32 KB
    const size_t SZ_X  = (size_t)B_ * T_ * D_ * sizeof(uint16_t);   // 64 MB
    const size_t SZ_XW = (size_t)T_ * B_ * G_ * sizeof(uint16_t);   // 256 MB

    uint16_t* Utf = (uint16_t*)(ws);
    uint16_t* Utb = (uint16_t*)(ws + SZ_WT);
    uint16_t* Wtf = (uint16_t*)(ws + 2 * SZ_WT);
    uint16_t* Wtb = (uint16_t*)(ws + 3 * SZ_WT);
    uint16_t* hF  = (uint16_t*)(ws + 4 * SZ_WT);
    uint16_t* hB  = (uint16_t*)(ws + 4 * SZ_WT + SZ_H);
    uint32_t* ctr = (uint32_t*)(ws + 4 * SZ_WT + 2 * SZ_H);
    uint16_t* xbf = (uint16_t*)(ws + 4 * SZ_WT + 2 * SZ_H + 256);
    uint16_t* xwF = (uint16_t*)((char*)xbf + SZ_X);
    uint16_t* xwB = (uint16_t*)((char*)xwF + SZ_XW);

    hipMemsetAsync(ctr, 0, 256, stream);

    init_pack<<<(G_ * D_) / 256, 256, 0, stream>>>(
        Wf, Uf, Wb, Ub, Wtf, Utf, Wtb, Utb, hF, hB);

    cvt_x<<<(B_ * T_ * D_) / (256 * 8), 256, 0, stream>>>(x, xbf);

    xw_gemm<<<dim3(T_ * 32, 2), 256, 0, stream>>>(
        xbf, Wtf, Wtb, bf, bb, xwF, xwB);

    lstm_rec<<<2 * NWG_, 512, 0, stream>>>(
        Utf, Utb, xwF, xwB, hF, hB, ctr, out);
}
